// HybridQuantumGNN_1314259992843
// MI455X (gfx1250) — compile-verified, matrix operand order checked by reading, not run on hardware
//
#include <hip/hip_runtime.h>

// ---------------------------------------------------------------------------
// HybridQuantumGNN for gfx1250 (MI455X).
// Dense node GEMMs -> v_wmma_f32_16x16x32_f16 (f32 accumulate).
// Edge softmax-aggregate -> 3-pass L2-atomic pipeline (workload is L2-bound:
// hw table (51MB) + edge lists fit in the 192MB L2).
// ---------------------------------------------------------------------------

typedef __attribute__((ext_vector_type(16))) _Float16 v16h;
typedef __attribute__((ext_vector_type(8)))  float    v8f;

#define NEG_SLOPE 0.2f
#define BN_EPS 1e-5f
#define GG 128                      // number of graphs
#define ENC_NEG_INF 0x007FFFFFu     // fenc(-inf)

// ---- orderable-uint encoding for float atomicMax -------------------------
__device__ __forceinline__ unsigned fenc(float f) {
  unsigned u = __float_as_uint(f);
  return (u & 0x80000000u) ? ~u : (u | 0x80000000u);
}
__device__ __forceinline__ float fdec(unsigned u) {
  return (u & 0x80000000u) ? __uint_as_float(u & 0x7FFFFFFFu)
                           : __uint_as_float(~u);
}

// ---- fills ----------------------------------------------------------------
__global__ void fill_f32_k(float* p, float v, size_t n) {
  size_t i = (size_t)blockIdx.x * blockDim.x + threadIdx.x;
  if (i < n) p[i] = v;
}
__global__ void fill_u32_k(unsigned* p, unsigned v, size_t n) {
  size_t i = (size_t)blockIdx.x * blockDim.x + threadIdx.x;
  if (i < n) p[i] = v;
}

// ---- f32 -> f16 conversions ----------------------------------------------
__global__ void cvt_f16_k(const float* in, _Float16* out, size_t n) {
  size_t i = (size_t)blockIdx.x * blockDim.x + threadIdx.x;
  if (i < n) out[i] = (_Float16)in[i];
}
// W is (K x Nout) row-major; store transposed (Nout x K) so B-tile lane loads
// are contiguous.
__global__ void cvt_w_f16_t_k(const float* W, _Float16* Wt, int K, int Nout) {
  int i = blockIdx.x * blockDim.x + threadIdx.x;
  if (i < K * Nout) {
    int k = i / Nout, n = i % Nout;
    Wt[(size_t)n * K + k] = (_Float16)W[i];
  }
}

// ---- WMMA GEMM: C[M x Nout] = A[M x K](f16) * Wt[Nout x K](f16)^T (+bias) --
// One wave per 16x16 output tile. 16-bit A layout (ISA 7.12.2): lanes 0-15
// hold row M=lane with K chunks [kk..kk+7],[kk+16..kk+23]; lanes 16-31 hold
// the same rows with chunks [kk+8..kk+15],[kk+24..kk+31]. B mirrors with
// lane = output column (a row of the pre-transposed Wt).
__global__ void wmma_gemm_k(const _Float16* __restrict__ A,
                            const _Float16* __restrict__ Bt,
                            const float* __restrict__ bias,
                            float* __restrict__ C,
                            int M, int K, int Nout) {
  int wave = (int)((blockIdx.x * blockDim.x + threadIdx.x) >> 5);
  int lane = threadIdx.x & 31;
  int tilesN = Nout >> 4;
  int tiles  = (M >> 4) * tilesN;
  if (wave >= tiles) return;          // wave-uniform: EXEC stays all-ones
  int tm = wave / tilesN;
  int tn = wave % tilesN;
  int half = lane >> 4;               // 0 or 1
  int l    = lane & 15;
  const _Float16* arow = A  + (size_t)(tm * 16 + l) * K + half * 8;
  const _Float16* brow = Bt + (size_t)(tn * 16 + l) * K + half * 8;

  v8f c = {};
  for (int kk = 0; kk < K; kk += 32) {
    union { v16h v; _Float16 e[16]; } a, b;
#pragma unroll
    for (int i = 0; i < 8; i++) {
      a.e[i]     = arow[kk + i];
      a.e[8 + i] = arow[kk + 16 + i];
      b.e[i]     = brow[kk + i];
      b.e[8 + i] = brow[kk + 16 + i];
    }
    c = __builtin_amdgcn_wmma_f32_16x16x32_f16(
        /*neg_a=*/false, a.v, /*neg_b=*/false, b.v,
        /*c_mod=*/(short)0, c, /*reuse_a=*/false, /*reuse_b=*/false);
  }

  int col = tn * 16 + l;
  float bv = bias ? bias[col] : 0.0f;
#pragma unroll
  for (int j = 0; j < 8; j++) {       // VGPR j: M = j (+8 for upper half)
    int row = tm * 16 + j + half * 8;
    C[(size_t)row * Nout + col] = c[j] + bv;
  }
}

// ---- per-node attention scalars s = hw . a_src, d = hw . a_dst -----------
__global__ void attn_scalars_k(const float* __restrict__ hw,
                               const float* __restrict__ asv,
                               const float* __restrict__ adv,
                               float* s, float* d, int N, int F) {
  int i = blockIdx.x * blockDim.x + threadIdx.x;
  if (i >= N) return;
  const float* hr = hw + (size_t)i * F;
  float ss = 0.f, dd = 0.f;
  for (int f = 0; f < F; f++) {
    float v = hr[f];
    ss += v * asv[f];
    dd += v * adv[f];
  }
  s[i] = ss;
  d[i] = dd;
}

// ---- edge pass 1: segment max of leaky logits ----------------------------
__global__ void edge_max_k(const int* __restrict__ ei, int E, int ET,
                           const float* __restrict__ s, const float* __restrict__ d,
                           unsigned* m) {
  int i = blockIdx.x * blockDim.x + threadIdx.x;
  if (i >= ET) return;
  int si, di;
  if (i < E) { si = ei[i]; di = ei[E + i]; } else { si = di = i - E; }
  float e = s[si] + d[di];
  e = (e >= 0.f) ? e : NEG_SLOPE * e;
  atomicMax(&m[di], fenc(e));
}

// ---- edge pass 2: segment sum of exp(e - m[dst]) -------------------------
__global__ void edge_sum_k(const int* __restrict__ ei, int E, int ET,
                           const float* __restrict__ s, const float* __restrict__ d,
                           const unsigned* __restrict__ m, float* denom) {
  int i = blockIdx.x * blockDim.x + threadIdx.x;
  if (i >= ET) return;
  int si, di;
  if (i < E) { si = ei[i]; di = ei[E + i]; } else { si = di = i - E; }
  float e = s[si] + d[di];
  e = (e >= 0.f) ? e : NEG_SLOPE * e;
  atomicAdd(&denom[di], __expf(e - fdec(m[di])));
}

// ---- edge pass 3: acc[dst] += alpha * hw[src] ----------------------------
__global__ void edge_aggr_k(const int* __restrict__ ei, int E, int ET,
                            const float* __restrict__ s, const float* __restrict__ d,
                            const unsigned* __restrict__ m,
                            const float* __restrict__ denom,
                            const float* __restrict__ hw,
                            float* acc, int F) {
  int i = blockIdx.x * blockDim.x + threadIdx.x;
  if (i >= ET) return;
  int si, di;
  if (i < E) { si = ei[i]; di = ei[E + i]; } else { si = di = i - E; }
  float e = s[si] + d[di];
  e = (e >= 0.f) ? e : NEG_SLOPE * e;
  float w = __expf(e - fdec(m[di])) / denom[di];
  const float* hr = hw + (size_t)si * F;
  float* ar = acc + (size_t)di * F;
  __builtin_prefetch(hr, 0, 0);   // global_prefetch_b8, row likely L2-resident
  for (int f = 0; f < F; f += 4) {
    float4 v = *(const float4*)(hr + f);
    atomicAdd(ar + f + 0, w * v.x);
    atomicAdd(ar + f + 1, w * v.y);
    atomicAdd(ar + f + 2, w * v.z);
    atomicAdd(ar + f + 3, w * v.w);
  }
}

// ---- batch-norm stats: per-feature sum / sum-of-squares ------------------
__global__ void bn_reduce_k(const float* __restrict__ h, int N, int F,
                            float* sum, float* sq) {
  int f = blockIdx.x;
  float s = 0.f, s2 = 0.f;
  for (int i = threadIdx.x; i < N; i += blockDim.x) {
    float v = h[(size_t)i * F + f];
    s += v;
    s2 += v * v;
  }
  __shared__ float ls[256], ls2[256];
  ls[threadIdx.x] = s;
  ls2[threadIdx.x] = s2;
  __syncthreads();
  for (int o = 128; o > 0; o >>= 1) {
    if ((int)threadIdx.x < o) {
      ls[threadIdx.x] += ls[threadIdx.x + o];
      ls2[threadIdx.x] += ls2[threadIdx.x + o];
    }
    __syncthreads();
  }
  if (threadIdx.x == 0) { sum[f] = ls[0]; sq[f] = ls2[0]; }
}

// ---- BN + ReLU + residual ------------------------------------------------
__global__ void bn_apply_k(const float* __restrict__ acc,
                           const float* __restrict__ resid,
                           const float* __restrict__ sum,
                           const float* __restrict__ sq,
                           const float* __restrict__ gamma,
                           const float* __restrict__ beta,
                           float* out, int N, int F) {
  size_t i = (size_t)blockIdx.x * blockDim.x + threadIdx.x;
  if (i >= (size_t)N * F) return;
  int f = (int)(i % F);
  float inv_n = 1.0f / (float)N;
  float mu = sum[f] * inv_n;
  float var = sq[f] * inv_n - mu * mu;
  float v = (acc[i] - mu) * rsqrtf(var + BN_EPS) * gamma[f] + beta[f];
  v = v > 0.f ? v : 0.f;
  out[i] = v + resid[i];
}

// ---- pooling -------------------------------------------------------------
__global__ void pool_counts_k(const int* batch, float* cnt, int N) {
  int i = blockIdx.x * blockDim.x + threadIdx.x;
  if (i < N) atomicAdd(&cnt[batch[i]], 1.0f);
}
__global__ void pool_accum_k(const float* __restrict__ h, const int* __restrict__ batch,
                             float* psum, unsigned* pmax, int N, int F) {
  size_t t = (size_t)blockIdx.x * blockDim.x + threadIdx.x;
  if (t >= (size_t)N * F) return;
  int i = (int)(t / F), f = (int)(t % F);
  int g = batch[i];
  float v = h[t];
  atomicAdd(&psum[(size_t)g * F + f], v);
  atomicMax(&pmax[(size_t)g * F + f], fenc(v));
}
__global__ void pool_final_k(const float* psum, const unsigned* pmax,
                             const float* cnt, float* hg, int F) {
  int t = blockIdx.x * blockDim.x + threadIdx.x;
  if (t >= GG * 2 * F) return;
  int g = t / (2 * F), f = t % (2 * F);
  hg[t] = (f < F) ? psum[(size_t)g * F + f] / cnt[g]
                  : fdec(pmax[(size_t)g * F + (f - F)]);
}

// ---- dense head: hq = tanh(hg @ Wq + bq); out = [hg,hq] @ Wr + br --------
__global__ void head_k(const float* __restrict__ hg,
                       const float* __restrict__ Wq, const float* __restrict__ bq,
                       const float* __restrict__ Wr, const float* __restrict__ br,
                       float* out) {
  __shared__ float row[264];
  int g = blockIdx.x, t = threadIdx.x;
  if (t < 256) row[t] = hg[(size_t)g * 256 + t];
  __syncthreads();
  if (t < 8) {
    float a = bq[t];
    for (int k = 0; k < 256; k++) a += row[k] * Wq[k * 8 + t];
    row[256 + t] = tanhf(a);
  }
  __syncthreads();
  if (t < 2) {
    float a = br[t];
    for (int k = 0; k < 264; k++) a += row[k] * Wr[k * 2 + t];
    out[g * 2 + t] = a;
  }
}

// ---------------------------------------------------------------------------
static inline unsigned grid1(size_t n, int b) { return (unsigned)((n + (size_t)b - 1) / (size_t)b); }

extern "C" void kernel_launch(void* const* d_in, const int* in_sizes, int n_in,
                              void* d_out, int out_size, void* d_ws, size_t ws_size,
                              hipStream_t stream) {
  (void)n_in; (void)out_size; (void)ws_size;

  const float* x     = (const float*)d_in[0];
  const int*   ei    = (const int*)d_in[1];
  const int*   batch = (const int*)d_in[2];
  const float* W0 = (const float*)d_in[3];
  const float* as0 = (const float*)d_in[4];
  const float* ad0 = (const float*)d_in[5];
  // b0/b1/b2 (d_in[6,12,18]) are mathematically cancelled by the following
  // BatchNorm (per-feature constant shift: mean absorbs it, var unchanged).
  const float* g0 = (const float*)d_in[7];
  const float* be0 = (const float*)d_in[8];
  const float* W1 = (const float*)d_in[9];
  const float* as1 = (const float*)d_in[10];
  const float* ad1 = (const float*)d_in[11];
  const float* g1 = (const float*)d_in[13];
  const float* be1 = (const float*)d_in[14];
  const float* W2 = (const float*)d_in[15];
  const float* as2 = (const float*)d_in[16];
  const float* ad2 = (const float*)d_in[17];
  const float* g2 = (const float*)d_in[19];
  const float* be2 = (const float*)d_in[20];
  const float* pW1 = (const float*)d_in[21];
  const float* pb1 = (const float*)d_in[22];
  const float* Wq = (const float*)d_in[23];
  const float* bq = (const float*)d_in[24];
  const float* Wr = (const float*)d_in[25];
  const float* br = (const float*)d_in[26];

  const int N  = in_sizes[0] / 64;  // 100000 (multiple of 16)
  const int E  = in_sizes[1] / 2;   // 1.6M
  const int ET = E + N;             // + self loops
  float* out = (float*)d_out;

  // ---- workspace carve-out (~231 MB) ----
  char* base = (char*)d_ws;
  size_t off = 0;
  auto take = [&](size_t bytes) -> void* {
    void* p = base + off;
    off += (bytes + 255) & ~(size_t)255;
    return p;
  };
  float*     h_cur = (float*)take((size_t)N * 128 * 4);
  float*     residb= (float*)take((size_t)N * 128 * 4);
  float*     hw    = (float*)take((size_t)N * 128 * 4);
  float*     acc   = (float*)take((size_t)N * 128 * 4);
  _Float16*  hbf   = (_Float16*)take((size_t)N * 128 * 2);
  _Float16*  wbf   = (_Float16*)take((size_t)128 * 128 * 2);
  float*     svec  = (float*)take((size_t)N * 4);
  float*     dvec  = (float*)take((size_t)N * 4);
  unsigned*  mvec  = (unsigned*)take((size_t)N * 4);
  float*     denom = (float*)take((size_t)N * 4);
  float*     bnsum = (float*)take(128 * 4);
  float*     bnsq  = (float*)take(128 * 4);
  float*     psum  = (float*)take((size_t)GG * 128 * 4);
  unsigned*  pmax  = (unsigned*)take((size_t)GG * 128 * 4);
  float*     cnt   = (float*)take(GG * 4);
  float*     hg    = (float*)take((size_t)GG * 256 * 4);

  const int B = 256;

  auto gemm = [&](const float* Hin, const float* W, const float* bias,
                  float* Cout, int Fin, int Fout) {
    cvt_f16_k<<<grid1((size_t)N * Fin, B), B, 0, stream>>>(Hin, hbf, (size_t)N * Fin);
    cvt_w_f16_t_k<<<grid1((size_t)Fin * Fout, B), B, 0, stream>>>(W, wbf, Fin, Fout);
    size_t tiles = (size_t)(N / 16) * (Fout / 16);
    wmma_gemm_k<<<grid1(tiles, 8), B, 0, stream>>>(hbf, wbf, bias, Cout, N, Fin, Fout);
  };

  auto gat_layer = [&](const float* Hin, int Fin, int Fout,
                       const float* W, const float* asv, const float* adv,
                       const float* gamma, const float* beta,
                       const float* resid_ptr) {
    gemm(Hin, W, nullptr, hw, Fin, Fout);
    attn_scalars_k<<<grid1(N, B), B, 0, stream>>>(hw, asv, adv, svec, dvec, N, Fout);
    fill_f32_k<<<grid1((size_t)N * Fout, B), B, 0, stream>>>(acc, 0.f, (size_t)N * Fout);
    fill_u32_k<<<grid1(N, B), B, 0, stream>>>(mvec, ENC_NEG_INF, N);
    fill_f32_k<<<grid1(N, B), B, 0, stream>>>(denom, 0.f, N);
    edge_max_k<<<grid1(ET, B), B, 0, stream>>>(ei, E, ET, svec, dvec, mvec);
    edge_sum_k<<<grid1(ET, B), B, 0, stream>>>(ei, E, ET, svec, dvec, mvec, denom);
    edge_aggr_k<<<grid1(ET, B), B, 0, stream>>>(ei, E, ET, svec, dvec, mvec, denom, hw, acc, Fout);
    bn_reduce_k<<<Fout, B, 0, stream>>>(acc, N, Fout, bnsum, bnsq);
    bn_apply_k<<<grid1((size_t)N * Fout, B), B, 0, stream>>>(
        acc, resid_ptr, bnsum, bnsq, gamma, beta, h_cur, N, Fout);
  };

  // layer 0: 64 -> 64, identity residual (= x)
  gat_layer(x, 64, 64, W0, as0, ad0, g0, be0, x);
  // layer 1: 64 -> 128, projected residual (h @ pW1 + pb1) before conv
  gemm(h_cur, pW1, pb1, residb, 64, 128);
  gat_layer(h_cur, 64, 128, W1, as1, ad1, g1, be1, residb);
  // layer 2: 128 -> 128, identity residual
  gat_layer(h_cur, 128, 128, W2, as2, ad2, g2, be2, h_cur);

  // pooling over graphs (mean || max)
  fill_f32_k<<<grid1((size_t)GG * 128, B), B, 0, stream>>>(psum, 0.f, (size_t)GG * 128);
  fill_u32_k<<<grid1((size_t)GG * 128, B), B, 0, stream>>>(pmax, ENC_NEG_INF, (size_t)GG * 128);
  fill_f32_k<<<grid1(GG, B), B, 0, stream>>>(cnt, 0.f, GG);
  pool_counts_k<<<grid1(N, B), B, 0, stream>>>(batch, cnt, N);
  pool_accum_k<<<grid1((size_t)N * 128, B), B, 0, stream>>>(h_cur, batch, psum, pmax, N, 128);
  pool_final_k<<<grid1((size_t)GG * 256, B), B, 0, stream>>>(psum, pmax, cnt, hg, 128);

  // quantum-surrogate + readout head
  head_k<<<GG, B, 0, stream>>>(hg, Wq, bq, Wr, br, out);
}
